// ODConv_9912784519526
// MI455X (gfx1250) — matrix operand order repaired and checked by reading, not compiled, not run on hardware
//
#include <hip/hip_runtime.h>
#include <stdint.h>

// ---------------- problem constants ----------------
#define N_    128
#define CIN   256
#define COUT  256
#define KS_   3
#define KEXP  4
#define DIM_  1024
#define DIMP  (DIM_ + 2)   // padded length (pad=1 each side)
#define R16_  16
#define BN_EPS 1e-5f

typedef __bf16 bf16_t;
typedef __attribute__((ext_vector_type(16))) __bf16 v16bf;
typedef __attribute__((ext_vector_type(8)))  __bf16 v8bf;
typedef __attribute__((ext_vector_type(8)))  float  v8f;
typedef unsigned int v4u_t __attribute__((ext_vector_type(4)));
typedef int          v4i_t __attribute__((ext_vector_type(4)));
typedef int          v8i_t __attribute__((ext_vector_type(8)));

union ABu { v16bf v; v8bf h[2]; };

#if defined(__HIP_DEVICE_COMPILE__) && __has_builtin(__builtin_amdgcn_tensor_load_to_lds) && __has_builtin(__builtin_amdgcn_s_wait_tensorcnt)
#define USE_TDM 1
#warning "CDNA5: TDM tensor_load_to_lds path ENABLED (device pass)"
#else
#define USE_TDM 0
#warning "CDNA5: TDM builtin NOT available in this pass; manual LDS staging"
#endif

// ---------------------------------------------------------------------------
// Kernel 1: fused x->bf16 conversion (with pad halo) + mean pooling.
// One block per (n, i) row: coalesced 4-float reads, LDS tree reduction.
// ---------------------------------------------------------------------------
__global__ void odconv_xmean_kernel(const float* __restrict__ x,
                                    bf16_t* __restrict__ xhp,
                                    float* __restrict__ mean)
{
    const int row = blockIdx.x;          // n*CIN + i
    const int t   = threadIdx.x;
    __shared__ float red[256];

    const float* src = x + (size_t)row * DIM_;
    bf16_t* dst = xhp + (size_t)row * DIMP + 1;

    float s = 0.f;
#pragma unroll
    for (int j = 0; j < DIM_ / 256; ++j) {
        float v = src[t + j * 256];
        s += v;
        dst[t + j * 256] = (bf16_t)v;
    }
    if (t == 0) {
        xhp[(size_t)row * DIMP]            = (bf16_t)0.f;
        xhp[(size_t)row * DIMP + DIMP - 1] = (bf16_t)0.f;
    }
    red[t] = s;
    __syncthreads();
#pragma unroll
    for (int off = 128; off > 0; off >>= 1) {
        if (t < off) red[t] += red[t + off];
        __syncthreads();
    }
    if (t == 0) mean[row] = red[0] * (1.0f / DIM_);
}

// ---------------------------------------------------------------------------
// Kernel 2: per-sample attention head from precomputed means.
// Produces: coef[n][4] = softmax(att4)*mtx, a1[n][3], a2[n][256],
//           scale[n][o] = att3[n][o]*gamma[o]/sqrt(var+eps), bias[o].
// ---------------------------------------------------------------------------
__global__ void odconv_attn_kernel(
    const float* __restrict__ mean,
    const float* __restrict__ attn_w, const float* __restrict__ attn_b,
    const float* __restrict__ fc1_w,  const float* __restrict__ fc1_b,
    const float* __restrict__ fc2_w,  const float* __restrict__ fc2_b,
    const float* __restrict__ fc3_w,  const float* __restrict__ fc3_b,
    const float* __restrict__ fc4_w,  const float* __restrict__ fc4_b,
    const float* __restrict__ mtx,
    const float* __restrict__ bn_g, const float* __restrict__ bn_b,
    const float* __restrict__ bn_m, const float* __restrict__ bn_v,
    float* __restrict__ coef, float* __restrict__ a1,
    float* __restrict__ a2,   float* __restrict__ scale,
    float* __restrict__ bias)
{
    __shared__ float mean_s[CIN];
    __shared__ float att_s[R16_];
    __shared__ float att4_s[KEXP];

    const int n = blockIdx.x;
    const int t = threadIdx.x;

    mean_s[t] = mean[(size_t)n * CIN + t];
    __syncthreads();

    if (t < R16_) {
        float acc = attn_b[t];
        const float* w = attn_w + t * CIN;
        for (int c = 0; c < CIN; ++c) acc += w[c] * mean_s[c];
        att_s[t] = fmaxf(acc, 0.f);
    }
    __syncthreads();

    if (t < KEXP) {
        float acc = fc4_b[t];
        for (int h = 0; h < R16_; ++h) acc += fc4_w[t * R16_ + h] * att_s[h];
        att4_s[t] = acc;
    }
    __syncthreads();

    if (t == 0) {
        float mx = att4_s[0];
        for (int k = 1; k < KEXP; ++k) mx = fmaxf(mx, att4_s[k]);
        float e[KEXP], den = 0.f;
        for (int k = 0; k < KEXP; ++k) { e[k] = __expf(att4_s[k] - mx); den += e[k]; }
        for (int k = 0; k < KEXP; ++k) coef[n * KEXP + k] = (e[k] / den) * mtx[k];
    }
    if (t < KS_) {
        float acc = fc1_b[t];
        for (int h = 0; h < R16_; ++h) acc += fc1_w[t * R16_ + h] * att_s[h];
        a1[n * KS_ + t] = 1.f / (1.f + __expf(-acc));
    }

    float acc2 = fc2_b[t], acc3 = fc3_b[t];
    for (int h = 0; h < R16_; ++h) {
        acc2 += fc2_w[t * R16_ + h] * att_s[h];
        acc3 += fc3_w[t * R16_ + h] * att_s[h];
    }
    a2[(size_t)n * CIN + t] = 1.f / (1.f + __expf(-acc2));
    float inv = bn_g[t] * rsqrtf(bn_v[t] + BN_EPS);
    scale[(size_t)n * COUT + t] = (1.f / (1.f + __expf(-acc3))) * inv;
    if (n == 0) bias[t] = bn_b[t] - bn_m[t] * inv;
}

// ---------------------------------------------------------------------------
// Kernel 3: single pass over the 402MB weight tensor. Folds the expert
// reduction (coef) and the i/s rank-1 scalings, emits bf16 Wh[n][s][o][i].
// ---------------------------------------------------------------------------
__global__ void odconv_fuse_kernel(
    const float* __restrict__ weight, const float* __restrict__ coef,
    const float* __restrict__ a1, const float* __restrict__ a2,
    bf16_t* __restrict__ Wh)
{
    const int n = blockIdx.x >> 8;
    const int o = blockIdx.x & 255;
    const int i = threadIdx.x;

    float cc[KEXP];
#pragma unroll
    for (int k = 0; k < KEXP; ++k) cc[k] = coef[n * KEXP + k];

    float acc[KS_] = {0.f, 0.f, 0.f};
#pragma unroll
    for (int k = 0; k < KEXP; ++k) {
        const float* wp = weight +
            ((((size_t)n * KEXP + k) * COUT + o) * CIN + i) * KS_;
        acc[0] += cc[k] * wp[0];
        acc[1] += cc[k] * wp[1];
        acc[2] += cc[k] * wp[2];
    }
    const float a2i = a2[(size_t)n * CIN + i];
#pragma unroll
    for (int s = 0; s < KS_; ++s) {
        float val = acc[s] * a2i * a1[n * KS_ + s];
        Wh[(((size_t)n * KS_ + s) * COUT + o) * CIN + i] = (bf16_t)val;
    }
}

// ---------------------------------------------------------------------------
// TDM: load a 2D tile (tile_d0 contiguous 2B elements x tile_d1 rows) from
// global into LDS with 16B padding after each 64B row (pad_interval code 3 =
// 16 dwords, pad_amount code 3 = 4 dwords) -> LDS row stride 40 bf16.
// Descriptor per CDNA5 ISA ch.8.
// ---------------------------------------------------------------------------
#if USE_TDM
__device__ __forceinline__ void tdm_load_tile_2d(
    uint32_t lds_addr, const void* gaddr,
    uint32_t tensor_d0, uint32_t tensor_d1,
    uint32_t tile_d0, uint32_t tile_d1, uint32_t stride0_elems)
{
    uint64_t ga = (uint64_t)gaddr;
    v4u_t g0;
    g0[0] = 1u;                                   // count=1, user descriptor
    g0[1] = lds_addr;                             // lds_addr [63:32]
    g0[2] = (uint32_t)(ga & 0xFFFFFFFFu);         // global_addr [95:64]
    g0[3] = (uint32_t)((ga >> 32) & 0x01FFFFFFu)  // global_addr [120:96]
          | (2u << 30);                           // type=2 ("image")
    v8i_t g1;
    g1[0] = (int)((1u << 16)                      // data_size = 2 bytes
                | (1u << 20)                      // pad_enable
                | (3u << 22)                      // pad_interval = 16 dwords
                | (3u << 25));                    // pad_amount  = 4 dwords
    g1[1] = (int)(tensor_d0 << 16);               // tensor_dim0 [79:48]
    g1[2] = (int)((tensor_d0 >> 16) | (tensor_d1 << 16)); // + tensor_dim1
    g1[3] = (int)((tensor_d1 >> 16) | (tile_d0 << 16));   // + tile_dim0
    g1[4] = (int)(tile_d1 & 0xFFFFu);             // tile_dim1 (tile_dim2=0)
    g1[5] = (int)stride0_elems;                   // tensor_dim0_stride lo
    g1[6] = 0;
    g1[7] = 0;
    v4i_t z = {0, 0, 0, 0};
#if __clang_major__ >= 23
    v8i_t z8 = {0, 0, 0, 0, 0, 0, 0, 0};
    __builtin_amdgcn_tensor_load_to_lds(g0, g1, z, z, z8, 0);
#else
    __builtin_amdgcn_tensor_load_to_lds(g0, g1, z, z, 0);
#endif
}
#endif

// ---------------------------------------------------------------------------
// Kernel 4: per-sample conv-as-GEMM with bf16 WMMA, double-buffered LDS,
// pipeline unrolled x2 so both LDS buffer bases are compile-time constants.
// out[n](256x1024) = sum_s Wh[n][s](256x256) @ xhp[n](256, shifted window)
// Block tile 128(o) x 128(t); 8 waves, each 32(o)x64(t) = 2x4 WMMA accums.
// A tile via TDM (fallback: b128 loads + LDS stores); B tile transposed in
// LDS so fragment gathers are aligned b128 ds loads. One barrier per K-step;
// next tile's TDM/global loads overlap current tile's 8 WMMAs.
// Epilogue folds att3*BN scale + BN shift + ReLU.
// ---------------------------------------------------------------------------
#define LDS_STRIDE 40   // 32 bf16 payload + 8 bf16 (16B) pad -> 80B rows
#define NIT (KS_ * (CIN / 32))   // 24 K-steps (even)

__global__ void odconv_conv_wmma_kernel(
    const bf16_t* __restrict__ Wh, const bf16_t* __restrict__ xhp,
    const float* __restrict__ scale, const float* __restrict__ bias,
    float* __restrict__ out)
{
    __shared__ bf16_t Asl[2][128 * LDS_STRIDE];   // [o_local][k_local]
    __shared__ bf16_t Bsl[2][128 * LDS_STRIDE];   // [t_local][k_local] (transposed)

    const int n     = blockIdx.z;
    const int oBase = blockIdx.y * 128;
    const int tBase = blockIdx.x * 128;

    const int tid  = threadIdx.x;
    const int wave = tid >> 5;
    const int lane = tid & 31;
    const int m    = lane & 15;
    const int half = lane >> 4;

    const int waveO = (wave >> 1) * 32;   // 0,32,64,96
    const int waveT = (wave & 1) * 64;    // 0,64

    const float* sc = scale + (size_t)n * COUT;
    __builtin_prefetch(sc + tid, 0, 1);           // -> global_prefetch_b8
    __builtin_prefetch(bias + tid, 0, 1);

    // ---- tile stager (A via TDM when available, B transposed manually) ----
    auto stage = [&](int buf, int it) {
        const int s  = it >> 3;        // 0..2
        const int kc = it & 7;         // 0..7
#if USE_TDM
        if (wave == 0) {
            const bf16_t* gsrc =
                Wh + (((size_t)n * KS_ + s) * COUT + oBase) * CIN + kc * 32;
            tdm_load_tile_2d((uint32_t)(uint64_t)(&Asl[buf][0]), gsrc,
                             /*tensor_d0=*/CIN, /*tensor_d1=*/COUT,
                             /*tile_d0=*/32, /*tile_d1=*/128,
                             /*stride0=*/CIN);
        }
#else
        {
            const int r  = tid >> 1;          // 0..127
            const int hs = (tid & 1) << 4;    // 0 or 16
            const v8bf* src = (const v8bf*)(Wh +
                (((size_t)n * KS_ + s) * COUT + oBase + r) * CIN + kc * 32 + hs);
            v8bf* dst = (v8bf*)(&Asl[buf][r * LDS_STRIDE + hs]);
            dst[0] = src[0];
            dst[1] = src[1];
        }
#endif
        {
            const int il   = tid >> 3;          // input-channel 0..31
            const int tseg = (tid & 7) << 4;    // 0,16,...,112
            const bf16_t* src = xhp +
                ((size_t)n * CIN + (size_t)(kc * 32 + il)) * DIMP +
                (size_t)(tBase + s + tseg);
            bf16_t* dst = &Bsl[buf][tseg * LDS_STRIDE + il];
#pragma unroll
            for (int j = 0; j < 16; ++j) dst[j * LDS_STRIDE] = src[j];
        }
    };

    v8f acc[2][4] = {};

    // ---- 8 WMMAs from buffer `buf` (call sites pass literal constants) ----
    auto compute = [&](int buf) {
        ABu afr[2], bfr[4];
#pragma unroll
        for (int fm = 0; fm < 2; ++fm) {
            const bf16_t* ap = &Asl[buf][(waveO + fm * 16 + m) * LDS_STRIDE];
            afr[fm].h[0] = *(const v8bf*)(ap + half * 8);        // K 0..7 / 8..15
            afr[fm].h[1] = *(const v8bf*)(ap + 16 + half * 8);   // K 16..23 / 24..31
        }
#pragma unroll
        for (int fn = 0; fn < 4; ++fn) {
            const bf16_t* bp =
                &Bsl[buf][(waveT + fn * 16 + m) * LDS_STRIDE + half * 8];
            bfr[fn].h[0] = *(const v8bf*)(bp);                   // K 0..7 / 8..15
            bfr[fn].h[1] = *(const v8bf*)(bp + 16);              // K 16..23 / 24..31
        }
#pragma unroll
        for (int fm = 0; fm < 2; ++fm)
#pragma unroll
            for (int fn = 0; fn < 4; ++fn)
                acc[fm][fn] = __builtin_amdgcn_wmma_f32_16x16x32_bf16(
                    false, afr[fm].v, false, bfr[fn].v,
                    (short)0, acc[fm][fn], false, false);
    };

    stage(0, 0);                       // preload first tile into buffer 0
#pragma unroll 1
    for (int it = 0; it < NIT; it += 2) {
        // ---- phase 0: compute buf0, stage buf1 ----
#if USE_TDM
        if (wave == 0) __builtin_amdgcn_s_wait_tensorcnt(0);
#endif
        __syncthreads();               // buf0 resident; buf1 free
        stage(1, it + 1);              // NIT even: it+1 always valid
        compute(0);

        // ---- phase 1: compute buf1, stage buf0 ----
#if USE_TDM
        if (wave == 0) __builtin_amdgcn_s_wait_tensorcnt(0);
#endif
        __syncthreads();               // buf1 resident; buf0 free
        if (it + 2 < NIT) stage(0, it + 2);
        compute(1);
    }

    // ---- epilogue: BN (scale folded with att3) + shift + ReLU ----
#pragma unroll
    for (int fm = 0; fm < 2; ++fm) {
        const int obase = oBase + waveO + fm * 16 + half * 8;
#pragma unroll
        for (int fn = 0; fn < 4; ++fn) {
            const int tg = tBase + waveT + fn * 16 + m;
            v8f c = acc[fm][fn];
#pragma unroll
            for (int v = 0; v < 8; ++v) {
                const int o = obase + v;
                float r = c[v] * sc[o] + bias[o];
                out[((size_t)n * COUT + o) * DIM_ + tg] = fmaxf(r, 0.f);
            }
        }
    }
}

// ---------------------------------------------------------------------------
// launcher
// ---------------------------------------------------------------------------
static inline size_t align256(size_t x) { return (x + 255) & ~(size_t)255; }

extern "C" void kernel_launch(void* const* d_in, const int* in_sizes, int n_in,
                              void* d_out, int out_size, void* d_ws, size_t ws_size,
                              hipStream_t stream)
{
    const float* x      = (const float*)d_in[0];
    const float* attn_w = (const float*)d_in[1];
    const float* attn_b = (const float*)d_in[2];
    const float* fc1_w  = (const float*)d_in[3];
    const float* fc1_b  = (const float*)d_in[4];
    const float* fc2_w  = (const float*)d_in[5];
    const float* fc2_b  = (const float*)d_in[6];
    const float* fc3_w  = (const float*)d_in[7];
    const float* fc3_b  = (const float*)d_in[8];
    const float* fc4_w  = (const float*)d_in[9];
    const float* fc4_b  = (const float*)d_in[10];
    const float* weight = (const float*)d_in[11];
    const float* mtx    = (const float*)d_in[12];
    const float* bn_g   = (const float*)d_in[13];
    const float* bn_b   = (const float*)d_in[14];
    const float* bn_m   = (const float*)d_in[15];
    const float* bn_v   = (const float*)d_in[16];

    char* ws = (char*)d_ws;
    size_t off = 0;
    float* coef  = (float*)(ws + off); off = align256(off + (size_t)N_ * KEXP * sizeof(float));
    float* a1    = (float*)(ws + off); off = align256(off + (size_t)N_ * KS_ * sizeof(float));
    float* a2    = (float*)(ws + off); off = align256(off + (size_t)N_ * CIN * sizeof(float));
    float* scale = (float*)(ws + off); off = align256(off + (size_t)N_ * COUT * sizeof(float));
    float* bias  = (float*)(ws + off); off = align256(off + (size_t)COUT * sizeof(float));
    float* mean  = (float*)(ws + off); off = align256(off + (size_t)N_ * CIN * sizeof(float));
    bf16_t* Wh   = (bf16_t*)(ws + off); off = align256(off + (size_t)N_ * KS_ * COUT * CIN * sizeof(bf16_t));
    bf16_t* xhp  = (bf16_t*)(ws + off); off = align256(off + (size_t)N_ * CIN * DIMP * sizeof(bf16_t));
    (void)ws_size; (void)in_sizes; (void)n_in; (void)out_size;

    // 1) x -> padded bf16 + per-(n,i) mean (coalesced; single pass over x)
    odconv_xmean_kernel<<<N_ * CIN, 256, 0, stream>>>(x, xhp, mean);

    // 2) attention head (tiny)
    odconv_attn_kernel<<<N_, 256, 0, stream>>>(
        mean, attn_w, attn_b, fc1_w, fc1_b, fc2_w, fc2_b, fc3_w, fc3_b,
        fc4_w, fc4_b, mtx, bn_g, bn_b, bn_m, bn_v,
        coef, a1, a2, scale, bias);

    // 3) one pass over the 402MB dynamic weight -> fused bf16 Wh
    odconv_fuse_kernel<<<N_ * COUT, 256, 0, stream>>>(weight, coef, a1, a2, Wh);

    // 4) WMMA conv-GEMM with folded BN+ReLU epilogue
    dim3 grid(DIM_ / 128, COUT / 128, N_);   // (8, 2, 128)
    odconv_conv_wmma_kernel<<<grid, 256, 0, stream>>>(Wh, xhp, scale, bias, (float*)d_out);
}
